// SelfAttentionNumberDimension_82016695485433
// MI455X (gfx1250) — matrix operand order repaired and checked by reading, not compiled, not run on hardware
//
#include <hip/hip_runtime.h>
#include <hip/hip_bf16.h>

// ---------------------------------------------------------------------------
// CDNA5 (gfx1250, wave32) bf16 WMMA implementation of 2D factored-bias causal
// self-attention.  All GEMMs use v_wmma_f32_16x16x32_bf16.
// GEMM inner loops are register double-buffered so next-iteration
// global_load_b128s overlap the current WMMAs (partial s_wait_loadcnt).
// ---------------------------------------------------------------------------

typedef __attribute__((ext_vector_type(16))) __bf16        v16bf;
typedef __attribute__((ext_vector_type(8)))  float         v8f;
typedef __attribute__((ext_vector_type(8)))  unsigned int  v8u;
typedef __attribute__((ext_vector_type(4)))  unsigned int  v4u;

#define B_    16
#define T_    1024
#define CIN_  512
#define H_    8
#define CPH_  64
#define HID_  512
#define NEGBIG (-1e9f)

// fp32 -> bf16 round-to-nearest-even (bit pattern as ushort)
static __device__ __forceinline__ unsigned short f2bf(float f) {
  unsigned int u = __float_as_uint(f);
  u += 0x7FFFu + ((u >> 16) & 1u);
  return (unsigned short)(u >> 16);
}

// 32 contiguous bytes (16 bf16) -> one operand register block
static __device__ __forceinline__ v8u ld32B(const unsigned short* p) {
  v4u a = *(const v4u*)p;
  v4u b = *(const v4u*)(p + 8);
  v8u r;
  r[0] = a[0]; r[1] = a[1]; r[2] = a[2]; r[3] = a[3];
  r[4] = b[0]; r[5] = b[1]; r[6] = b[2]; r[7] = b[3];
  return r;
}

// A-fragment load for 16x32 bf16 A matrix (row-major source row pointer).
// elements 0..15 <- row[2*k0a .. 2*k0a+15]  (K = 2*k0a..2*k0a+15)
// with k0a = (lane>=16 ? 8 : 0), contiguous K order matching the B frags.
static __device__ __forceinline__ v8u ldA(const unsigned short* prow, int k0a) {
  v4u a = *(const v4u*)(prow + 2 * k0a);
  v4u b = *(const v4u*)(prow + 2 * k0a + 8);
  v8u r;
  r[0] = a[0]; r[1] = a[1]; r[2] = a[2]; r[3] = a[3];
  r[4] = b[0]; r[5] = b[1]; r[6] = b[2]; r[7] = b[3];
  return r;
}

static __device__ __forceinline__ v16bf asbf(v8u r) {
  return __builtin_bit_cast(v16bf, r);
}

static __device__ __forceinline__ v8f wmma_bf16(v16bf a, v16bf b, v8f c) {
  return __builtin_amdgcn_wmma_f32_16x16x32_bf16(
      /*neg_a=*/false, a, /*neg_b=*/false, b,
      /*c_mod=*/(short)0, c, /*reuse_a=*/false, /*reuse_b=*/false);
}

// ---------------------------------------------------------------------------
// Kernel 1: fp32 -> bf16 elementwise convert (x)
// ---------------------------------------------------------------------------
__global__ void conv_x_kernel(const float* __restrict__ x,
                              unsigned short* __restrict__ xb, int n) {
  int i = blockIdx.x * blockDim.x + threadIdx.x;
  if (i < n) xb[i] = f2bf(x[i]);
}

// ---------------------------------------------------------------------------
// Kernel 2: 512x512 transpose + bf16 convert (weights -> B^T operand layout)
// dst[r][c] = src[c*512 + r]
// ---------------------------------------------------------------------------
__global__ void transpose512_kernel(const float* __restrict__ src,
                                    unsigned short* __restrict__ dst) {
  int i = blockIdx.x * blockDim.x + threadIdx.x;  // dst linear index
  int r = i >> 9, c = i & 511;
  dst[i] = f2bf(src[c * 512 + r]);
}

// ---------------------------------------------------------------------------
// Kernel 3: QKV projection GEMM.  C[16384 x 512] = xb[16384 x 512] * W
// blockIdx.z selects Wq/Wk/Wv; epilogue scatters into attention layouts:
//   Q  -> [B,H,T,CPH] bf16, pre-scaled by CPH^-0.5 = 0.125
//   K  -> [B,H,T,CPH] bf16  (serves directly as B^T for Q*K^T)
//   V  -> [B,H,CPH,T] bf16  (transposed: serves as B^T for P*V)
// 128 threads = 4 waves, 64x64 tile, each wave owns 16 rows x 64 cols.
// Inner loop double-buffered: loads for kk+32 issue before WMMAs of kk.
// ---------------------------------------------------------------------------
__global__ __launch_bounds__(128)
void qkv_gemm_kernel(const unsigned short* __restrict__ xb,
                     const unsigned short* __restrict__ WqT,
                     const unsigned short* __restrict__ WkT,
                     const unsigned short* __restrict__ WvT,
                     unsigned short* __restrict__ Qb,
                     unsigned short* __restrict__ Kb,
                     unsigned short* __restrict__ VTb) {
  const int z    = blockIdx.z;
  const unsigned short* Bt = (z == 0) ? WqT : (z == 1) ? WkT : WvT;
  const int w    = threadIdx.x >> 5;
  const int lane = threadIdx.x & 31;
  const int sub  = lane >> 4;       // lane half
  const int lid  = lane & 15;
  const int k0a  = sub * 8;         // A-frag K base within 32-chunk
  const int khB  = sub * 16;        // B-frag K base within 32-chunk
  const int m0   = blockIdx.y * 64 + w * 16;
  const int n0   = blockIdx.x * 64;

  const unsigned short* arow = xb + (size_t)(m0 + lid) * CIN_;
  const unsigned short* bp0  = Bt + (size_t)(n0 +  0 + lid) * CIN_ + khB;
  const unsigned short* bp1  = Bt + (size_t)(n0 + 16 + lid) * CIN_ + khB;
  const unsigned short* bp2  = Bt + (size_t)(n0 + 32 + lid) * CIN_ + khB;
  const unsigned short* bp3  = Bt + (size_t)(n0 + 48 + lid) * CIN_ + khB;

  v8f c0 = {}, c1 = {}, c2 = {}, c3 = {};

  // prologue: fragments for kk = 0
  v8u aF  = ldA(arow, k0a);
  v8u bF0 = ld32B(bp0), bF1 = ld32B(bp1), bF2 = ld32B(bp2), bF3 = ld32B(bp3);

  for (int kk = 32; kk < CIN_; kk += 32) {
    // issue next-iteration loads first so they overlap the WMMAs below
    v8u aN  = ldA(arow + kk, k0a);
    v8u bN0 = ld32B(bp0 + kk), bN1 = ld32B(bp1 + kk);
    v8u bN2 = ld32B(bp2 + kk), bN3 = ld32B(bp3 + kk);
    __builtin_prefetch(arow + kk + 32, 0, 1);  // global_prefetch_b8

    c0 = wmma_bf16(asbf(aF), asbf(bF0), c0);
    c1 = wmma_bf16(asbf(aF), asbf(bF1), c1);
    c2 = wmma_bf16(asbf(aF), asbf(bF2), c2);
    c3 = wmma_bf16(asbf(aF), asbf(bF3), c3);

    aF = aN; bF0 = bN0; bF1 = bN1; bF2 = bN2; bF3 = bN3;
  }
  // epilogue iteration
  c0 = wmma_bf16(asbf(aF), asbf(bF0), c0);
  c1 = wmma_bf16(asbf(aF), asbf(bF1), c1);
  c2 = wmma_bf16(asbf(aF), asbf(bF2), c2);
  c3 = wmma_bf16(asbf(aF), asbf(bF3), c3);

  v8f cs[4] = {c0, c1, c2, c3};
#pragma unroll
  for (int i = 0; i < 4; ++i) {
#pragma unroll
    for (int j = 0; j < 8; ++j) {
      const int gm = m0 + j + sub * 8;           // C layout: row = vgpr + 8*half
      const int gn = n0 + i * 16 + lid;
      const int b  = gm >> 10, t = gm & 1023;
      const int h  = gn >> 6,  d = gn & 63;
      const float val = cs[i][j];
      if (z == 0)
        Qb[(((size_t)(b * H_ + h) * T_) + t) * CPH_ + d] = f2bf(val * 0.125f);
      else if (z == 1)
        Kb[(((size_t)(b * H_ + h) * T_) + t) * CPH_ + d] = f2bf(val);
      else
        VTb[(((size_t)(b * H_ + h) * CPH_) + d) * T_ + t] = f2bf(val);
    }
  }
}

// ---------------------------------------------------------------------------
// Kernel 4: fused flash-attention with factorized ND relative bias + causal
// mask.  One wave (32 threads) per (batch, head, 16-row q tile).
// Online softmax; S and O share the C-fragment row mapping so rescale is
// elementwise.  P goes through a 1KB LDS tile to convert C-layout -> A-layout.
// ---------------------------------------------------------------------------
__global__ __launch_bounds__(32)
void attn_kernel(const unsigned short* __restrict__ Qb,
                 const unsigned short* __restrict__ Kb,
                 const unsigned short* __restrict__ VTb,
                 const float* __restrict__ rb0,
                 const float* __restrict__ rb1,
                 unsigned short* __restrict__ Ob) {
  __shared__ float rb0s[64];
  __shared__ float rb1s[64];
  __shared__ __align__(16) unsigned short ptile[16][32];

  const int lane = threadIdx.x;
  const int qt   = blockIdx.x;   // q tile (16 rows)
  const int h    = blockIdx.y;
  const int b    = blockIdx.z;

  for (int i = lane; i < 64; i += 32) {
    rb0s[i] = rb0[h * 64 + i];
    rb1s[i] = rb1[h * 64 + i];
  }
  __syncthreads();

  const int sub = lane >> 4;
  const int lid = lane & 15;
  const int k0a = sub * 8;
  const int khB = sub * 16;
  const int q0  = qt * 16;
  const int bh  = b * H_ + h;

  // Q A-fragments for this wave's 16 rows (contract d = 0..31 and 32..63)
  const unsigned short* qrow = Qb + ((size_t)bh * T_ + q0 + lid) * CPH_;
  const v16bf qa0 = asbf(ldA(qrow,      k0a));
  const v16bf qa1 = asbf(ldA(qrow + 32, k0a));

  float m_i[8], l_i[8];
  v8f o0 = {}, o1 = {}, o2 = {}, o3 = {};
#pragma unroll
  for (int j = 0; j < 8; ++j) { m_i[j] = -1e30f; l_i[j] = 0.f; }

  const int nkc = q0 / 32 + 1;  // causal: only chunks with k0 <= q0+15
  for (int kc = 0; kc < nkc; ++kc) {
    const int k0 = kc * 32;

    // ---- S[16 x 32] = Q * K^T : two 16x16 C frags, 4 WMMAs ----
    v8f s0 = {}, s1 = {};
    {
      const unsigned short* kr0 = Kb + ((size_t)bh * T_ + k0 +      lid) * CPH_ + khB;
      const unsigned short* kr1 = Kb + ((size_t)bh * T_ + k0 + 16 + lid) * CPH_ + khB;
      v8u kb00 = ld32B(kr0);
      v8u kb01 = ld32B(kr0 + 32);
      v8u kb10 = ld32B(kr1);
      v8u kb11 = ld32B(kr1 + 32);
      s0 = wmma_bf16(qa0, asbf(kb00), s0);
      s0 = wmma_bf16(qa1, asbf(kb01), s0);
      s1 = wmma_bf16(qa0, asbf(kb10), s1);
      s1 = wmma_bf16(qa1, asbf(kb11), s1);
    }

    // ---- bias + causal mask + per-row max ----
    float mx[8];
#pragma unroll
    for (int j = 0; j < 8; ++j) {
      const int q  = q0 + j + sub * 8;
      const int kA = k0 + lid;
      const int kB = k0 + 16 + lid;
      const float bias0 = rb0s[31 + (kA >> 5) - (q >> 5)] + rb1s[31 + (kA & 31) - (q & 31)];
      const float bias1 = rb0s[31 + (kB >> 5) - (q >> 5)] + rb1s[31 + (kB & 31) - (q & 31)];
      float v0 = s0[j] + bias0; if (kA > q) v0 = NEGBIG;
      float v1 = s1[j] + bias1; if (kB > q) v1 = NEGBIG;
      s0[j] = v0; s1[j] = v1;
      mx[j] = fmaxf(v0, v1);
    }
#pragma unroll
    for (int m = 1; m < 16; m <<= 1)
#pragma unroll
      for (int j = 0; j < 8; ++j)
        mx[j] = fmaxf(mx[j], __shfl_xor(mx[j], m, 32));

    // ---- online softmax update; stash P (bf16) into LDS in A-layout feed ----
    float alpha[8], rs[8];
#pragma unroll
    for (int j = 0; j < 8; ++j) {
      const float mnew = fmaxf(m_i[j], mx[j]);
      alpha[j] = __expf(m_i[j] - mnew);
      const float p0 = __expf(s0[j] - mnew);
      const float p1 = __expf(s1[j] - mnew);
      m_i[j] = mnew;
      rs[j]  = p0 + p1;
      const int row = j + sub * 8;
      ptile[row][lid]      = f2bf(p0);
      ptile[row][16 + lid] = f2bf(p1);
    }
#pragma unroll
    for (int m = 1; m < 16; m <<= 1)
#pragma unroll
      for (int j = 0; j < 8; ++j)
        rs[j] += __shfl_xor(rs[j], m, 32);
#pragma unroll
    for (int j = 0; j < 8; ++j) {
      l_i[j] = l_i[j] * alpha[j] + rs[j];
      o0[j] *= alpha[j];
      o1[j] *= alpha[j];
      o2[j] *= alpha[j];
      o3[j] *= alpha[j];
    }

    __syncthreads();  // single-wave WG: S_NOP barrier, but fences DS counters

    // ---- O += P * V : P as A-frag (16x32), V^T rows as B-frags, 4 WMMAs ----
    const v16bf pa = asbf(ldA(&ptile[lid][0], k0a));
    const unsigned short* vbase = VTb + (size_t)bh * CPH_ * T_ + k0 + khB;
    v8u vb0 = ld32B(vbase + (size_t)( 0 + lid) * T_);
    v8u vb1 = ld32B(vbase + (size_t)(16 + lid) * T_);
    v8u vb2 = ld32B(vbase + (size_t)(32 + lid) * T_);
    v8u vb3 = ld32B(vbase + (size_t)(48 + lid) * T_);
    o0 = wmma_bf16(pa, asbf(vb0), o0);
    o1 = wmma_bf16(pa, asbf(vb1), o1);
    o2 = wmma_bf16(pa, asbf(vb2), o2);
    o3 = wmma_bf16(pa, asbf(vb3), o3);

    __syncthreads();  // protect ptile before next chunk overwrites it
  }

  // ---- finalize: O /= l, write bf16 out[b, t, h*64+d] (A operand of Wo GEMM)
#pragma unroll
  for (int j = 0; j < 8; ++j) {
    const float inv = 1.0f / l_i[j];
    const int q = q0 + j + sub * 8;
    unsigned short* orow = Ob + ((size_t)b * T_ + q) * (H_ * CPH_) + h * CPH_;
    orow[ 0 + lid] = f2bf(o0[j] * inv);
    orow[16 + lid] = f2bf(o1[j] * inv);
    orow[32 + lid] = f2bf(o2[j] * inv);
    orow[48 + lid] = f2bf(o3[j] * inv);
  }
}

// ---------------------------------------------------------------------------
// Kernel 5: output projection.  y[16384 x 512] f32 = Ob[16384 x 512] * Wo
// Same double-buffered structure as the QKV GEMM; writes fp32 to d_out.
// ---------------------------------------------------------------------------
__global__ __launch_bounds__(128)
void out_gemm_kernel(const unsigned short* __restrict__ Ab,
                     const unsigned short* __restrict__ WoT,
                     float* __restrict__ y) {
  const int w    = threadIdx.x >> 5;
  const int lane = threadIdx.x & 31;
  const int sub  = lane >> 4;
  const int lid  = lane & 15;
  const int k0a  = sub * 8;
  const int khB  = sub * 16;
  const int m0   = blockIdx.y * 64 + w * 16;
  const int n0   = blockIdx.x * 64;

  const unsigned short* arow = Ab + (size_t)(m0 + lid) * HID_;
  const unsigned short* bp0  = WoT + (size_t)(n0 +  0 + lid) * HID_ + khB;
  const unsigned short* bp1  = WoT + (size_t)(n0 + 16 + lid) * HID_ + khB;
  const unsigned short* bp2  = WoT + (size_t)(n0 + 32 + lid) * HID_ + khB;
  const unsigned short* bp3  = WoT + (size_t)(n0 + 48 + lid) * HID_ + khB;

  v8f c0 = {}, c1 = {}, c2 = {}, c3 = {};

  v8u aF  = ldA(arow, k0a);
  v8u bF0 = ld32B(bp0), bF1 = ld32B(bp1), bF2 = ld32B(bp2), bF3 = ld32B(bp3);

  for (int kk = 32; kk < HID_; kk += 32) {
    v8u aN  = ldA(arow + kk, k0a);
    v8u bN0 = ld32B(bp0 + kk), bN1 = ld32B(bp1 + kk);
    v8u bN2 = ld32B(bp2 + kk), bN3 = ld32B(bp3 + kk);
    __builtin_prefetch(arow + kk + 32, 0, 1);

    c0 = wmma_bf16(asbf(aF), asbf(bF0), c0);
    c1 = wmma_bf16(asbf(aF), asbf(bF1), c1);
    c2 = wmma_bf16(asbf(aF), asbf(bF2), c2);
    c3 = wmma_bf16(asbf(aF), asbf(bF3), c3);

    aF = aN; bF0 = bN0; bF1 = bN1; bF2 = bN2; bF3 = bN3;
  }
  c0 = wmma_bf16(asbf(aF), asbf(bF0), c0);
  c1 = wmma_bf16(asbf(aF), asbf(bF1), c1);
  c2 = wmma_bf16(asbf(aF), asbf(bF2), c2);
  c3 = wmma_bf16(asbf(aF), asbf(bF3), c3);

  v8f cs[4] = {c0, c1, c2, c3};
#pragma unroll
  for (int i = 0; i < 4; ++i) {
#pragma unroll
    for (int j = 0; j < 8; ++j) {
      const int gm = m0 + j + sub * 8;
      const int gn = n0 + i * 16 + lid;
      y[(size_t)gm * HID_ + gn] = cs[i][j];
    }
  }
}

// ---------------------------------------------------------------------------
// Host launcher
// ---------------------------------------------------------------------------
extern "C" void kernel_launch(void* const* d_in, const int* in_sizes, int n_in,
                              void* d_out, int out_size, void* d_ws, size_t ws_size,
                              hipStream_t stream) {
  (void)in_sizes; (void)n_in; (void)out_size; (void)ws_size;
  const float* x   = (const float*)d_in[0];
  const float* Wq  = (const float*)d_in[1];
  const float* Wk  = (const float*)d_in[2];
  const float* Wv  = (const float*)d_in[3];
  const float* Wo  = (const float*)d_in[4];
  const float* rb0 = (const float*)d_in[5];
  const float* rb1 = (const float*)d_in[6];
  float* y = (float*)d_out;

  char* ws = (char*)d_ws;
  size_t off = 0;
  auto carve = [&](size_t bytes) -> void* {
    void* p = (void*)(ws + off);
    off += (bytes + 255) & ~(size_t)255;
    return p;
  };

  const size_t MT = (size_t)B_ * T_;                   // 16384 rows
  unsigned short* xb  = (unsigned short*)carve(MT * CIN_ * 2);       // 16 MB
  unsigned short* WqT = (unsigned short*)carve((size_t)512 * 512 * 2);
  unsigned short* WkT = (unsigned short*)carve((size_t)512 * 512 * 2);
  unsigned short* WvT = (unsigned short*)carve((size_t)512 * 512 * 2);
  unsigned short* WoT = (unsigned short*)carve((size_t)512 * 512 * 2);
  unsigned short* Qb  = (unsigned short*)carve((size_t)B_ * H_ * T_ * CPH_ * 2);  // 16 MB
  unsigned short* Kb  = (unsigned short*)carve((size_t)B_ * H_ * T_ * CPH_ * 2);  // 16 MB
  unsigned short* VTb = (unsigned short*)carve((size_t)B_ * H_ * CPH_ * T_ * 2);  // 16 MB
  unsigned short* Ob  = (unsigned short*)carve(MT * (size_t)(H_ * CPH_) * 2);     // 16 MB

  // 1) convert x, transpose-convert weights to B^T bf16 operand layouts
  conv_x_kernel<<<(int)(MT * CIN_ / 256), 256, 0, stream>>>(x, xb, (int)(MT * CIN_));
  transpose512_kernel<<<1024, 256, 0, stream>>>(Wq, WqT);
  transpose512_kernel<<<1024, 256, 0, stream>>>(Wk, WkT);
  transpose512_kernel<<<1024, 256, 0, stream>>>(Wv, WvT);
  transpose512_kernel<<<1024, 256, 0, stream>>>(Wo, WoT);

  // 2) QKV projections (z = 0/1/2 selects Wq/Wk/Wv and the output layout)
  qkv_gemm_kernel<<<dim3(HID_ / 64, (unsigned)(MT / 64), 3), 128, 0, stream>>>(
      xb, WqT, WkT, WvT, Qb, Kb, VTb);

  // 3) fused flash attention with ND relative bias + causal mask
  attn_kernel<<<dim3(T_ / 16, H_, B_), 32, 0, stream>>>(Qb, Kb, VTb, rb0, rb1, Ob);

  // 4) output projection -> fp32 d_out
  out_gemm_kernel<<<dim3(HID_ / 64, (unsigned)(MT / 64)), 128, 0, stream>>>(Ob, WoT, y);
}
